// GNN_64424509440778
// MI455X (gfx1250) — compile-verified, matrix operand order checked by reading, not run on hardware
//
#include <hip/hip_runtime.h>
#include <math.h>

typedef _Float16 h16;
typedef _Float16 v16h __attribute__((ext_vector_type(16)));
typedef _Float16 v8h  __attribute__((ext_vector_type(8)));
typedef float    v8f  __attribute__((ext_vector_type(8)));

#define HDIM 128
#define RBF_BINS 64

// ---------------------------------------------------------------------------
// WMMA helpers (CDNA5 gfx1250, wave32)
// ---------------------------------------------------------------------------
static __device__ __forceinline__ v8f wmma16(v16h a, v16h b, v8f c) {
  // D = A(16x32 f16) x B(32x16 f16) + C(16x16 f32)
  return __builtin_amdgcn_wmma_f32_16x16x32_f16(false, a, false, b, (short)0, c,
                                                false, false);
}

static __device__ __forceinline__ float elu_f(float x) {
  return x > 0.f ? x : expm1f(x);
}

// A-fragment from a row-major f16 LDS tile (lda in halves).
// ISA layout (16-bit A 16x32): lane<16 -> K = kb..kb+7 (VGPR0-3), kb+16..kb+23
// (VGPR4-7), kb = (lane>=16)*8.
static __device__ __forceinline__ v16h load_a_frag(const h16* t, int lda,
                                                   int kt, int lane) {
  int row = lane & 15;
  int kb  = (lane >> 4) << 3;
  const h16* p = t + row * lda + kt * 32 + kb;
  v8h lo = *(const v8h*)p;         // K kb..kb+7      -> ds_read_b128
  v8h hi = *(const v8h*)(p + 16);  // K kb+16..kb+23  -> ds_read_b128
  v16h a;
#pragma unroll
  for (int q = 0; q < 8; ++q) { a[q] = lo[q]; a[8 + q] = hi[q]; }
  return a;
}

// B-fragment from weight buffer pre-packed fragment-contiguous:
// lane's 16 halves are 32 contiguous bytes -> 2x global_load_b128.
static __device__ __forceinline__ v16h load_b_frag(const h16* pk, int NT,
                                                   int kt, int nt, int lane) {
  return *(const v16h*)(pk + (((size_t)(kt * NT + nt)) * 32 + lane) * 16);
}

// In-register LayerNorm + ELU over a 16x128 tile held as 8 f32 C-tiles.
// Row stats reduced across the 16-lane half-group with shfl_xor 1/2/4/8.
static __device__ __forceinline__ void ln_elu(v8f (&acc)[8], const float* g,
                                              const float* b, int lane) {
  int col = lane & 15;
  float gv[8], bv[8];
#pragma unroll
  for (int nt = 0; nt < 8; ++nt) { gv[nt] = g[nt * 16 + col]; bv[nt] = b[nt * 16 + col]; }
#pragma unroll
  for (int r = 0; r < 8; ++r) {
    float s = 0.f, q = 0.f;
#pragma unroll
    for (int nt = 0; nt < 8; ++nt) { float v = acc[nt][r]; s += v; q += v * v; }
#pragma unroll
    for (int m = 1; m <= 8; m <<= 1) { s += __shfl_xor(s, m, 32); q += __shfl_xor(q, m, 32); }
    float mean = s * (1.f / 128.f);
    float var  = q * (1.f / 128.f) - mean * mean;
    float rstd = rsqrtf(var + 1e-5f);
#pragma unroll
    for (int nt = 0; nt < 8; ++nt) {
      float v = (acc[nt][r] - mean) * rstd * gv[nt] + bv[nt];
      acc[nt][r] = elu_f(v);
    }
  }
}

// C-tiles -> f16 row-major LDS tile (for use as next GEMM's A matrix).
static __device__ __forceinline__ void c_to_lds(const v8f (&acc)[8], h16* t,
                                                int lda, int lane) {
  int col  = lane & 15;
  int mofs = (lane >> 4) << 3;
#pragma unroll
  for (int nt = 0; nt < 8; ++nt)
#pragma unroll
    for (int r = 0; r < 8; ++r)
      t[(r + mofs) * lda + nt * 16 + col] = (h16)acc[nt][r];
}

// ---------------------------------------------------------------------------
// Utility kernels
// ---------------------------------------------------------------------------
__global__ void zero_f32_kernel(float* p, int n) {
  int t = blockIdx.x * blockDim.x + threadIdx.x;
  if (t < n) p[t] = 0.f;
}

__global__ void pack_b_kernel(const float* __restrict__ W, h16* __restrict__ out,
                              int KT, int NT, int N) {
  int t = blockIdx.x * blockDim.x + threadIdx.x;
  if (t >= KT * NT * 512) return;
  int q = t & 15, lane = (t >> 4) & 31, f = t >> 9;
  int nt = f % NT, kt = f / NT;
  int col = lane & 15, kb = (lane >> 4) << 3;
  int k = kt * 32 + kb + (q < 8 ? q : q + 8);   // K pattern mirrors A layout
  out[t] = (h16)W[(size_t)k * N + nt * 16 + col];
}

__global__ void embed_kernel(const float* __restrict__ x,
                             const float* __restrict__ Wemb,
                             float* __restrict__ xf, h16* __restrict__ xh, int N) {
  int t = blockIdx.x * blockDim.x + threadIdx.x;
  if (t >= N * HDIM) return;
  float v = x[t >> 7] * Wemb[t & 127];
  xf[t] = v;
  xh[t] = (h16)v;
}

__global__ void count_edge_kernel(const int* __restrict__ ei, float* cnt, int E) {
  int t = blockIdx.x * blockDim.x + threadIdx.x;
  if (t < E) atomicAdd(&cnt[ei[t]], 1.f);
}

__global__ void clamp1_kernel(float* cnt, int n) {
  int t = blockIdx.x * blockDim.x + threadIdx.x;
  if (t < n) cnt[t] = fmaxf(cnt[t], 1.f);
}

__global__ void count_node_kernel(const int* __restrict__ batch, float* gcnt, int N) {
  int t = blockIdx.x * blockDim.x + threadIdx.x;
  if (t < N) atomicAdd(&gcnt[batch[t]], 1.f);
}

// ---------------------------------------------------------------------------
// RBF expansion + edge embedding:  eemb = elu(rbf @ W_ee + b_ee)   [E,128] f16
// ---------------------------------------------------------------------------
__global__ __launch_bounds__(128) void rbf_edge_kernel(
    const float* __restrict__ ea, const h16* __restrict__ pWee,
    const float* __restrict__ bee, h16* __restrict__ eemb, int E) {
  __shared__ h16 sm[4][16 * RBF_BINS];
  int lane = threadIdx.x & 31, w = threadIdx.x >> 5;
  int base = (blockIdx.x * 4 + w) * 16;
  h16* A = sm[w];
  {
    int row = lane & 15;
    int k0  = (lane >> 4) * 32;
    int e = base + row; if (e >= E) e = E - 1;
    float d = ea[e];
    const float step  = 10.0f / 63.0f;
    const float gamma = 1.0f / (step * step);
#pragma unroll
    for (int k = 0; k < 32; ++k) {
      float dd = d - (float)(k0 + k) * step;
      A[row * RBF_BINS + k0 + k] = (h16)__expf(-gamma * dd * dd);
    }
  }
  __syncthreads();
  v8f acc[8] = {};
#pragma unroll
  for (int kt = 0; kt < 2; ++kt) {
    v16h a = load_a_frag(A, RBF_BINS, kt, lane);
#pragma unroll
    for (int nt = 0; nt < 8; ++nt)
      acc[nt] = wmma16(a, load_b_frag(pWee, 8, kt, nt, lane), acc[nt]);
  }
  int col = lane & 15, mofs = (lane >> 4) << 3;
#pragma unroll
  for (int nt = 0; nt < 8; ++nt) {
    float bv = bee[nt * 16 + col];
#pragma unroll
    for (int r = 0; r < 8; ++r) {
      int e = base + r + mofs;
      if (e < E)
        eemb[(size_t)e * HDIM + nt * 16 + col] = (h16)elu_f(acc[nt][r] + bv);
    }
  }
}

// ---------------------------------------------------------------------------
// Edge MLP + attention gate + scatter-add, M-blocked at 32 edges/wave so each
// B fragment load feeds two WMMAs (halves WGP$->VGPR weight traffic).
// m = elu(LN(concat[x_i,x_j,e] @ We1)); m = elu(LN(m @ We2)); m *= (m@wa+ba)
// macc[i] += m   (f32 atomics; node kernel divides by cnt)
// ---------------------------------------------------------------------------
__global__ __launch_bounds__(128) void edge_mlp_kernel(
    const int* __restrict__ ei, const int* __restrict__ ej,
    const h16* __restrict__ xh, const h16* __restrict__ eemb,
    const h16* __restrict__ pWe1, const h16* __restrict__ pWe2,
    const float* __restrict__ g1, const float* __restrict__ b1,
    const float* __restrict__ g2, const float* __restrict__ b2,
    const float* __restrict__ wa, const float* __restrict__ ba,
    float* __restrict__ macc, int E) {
  __shared__ h16 sm[4][32 * 384];      // 96 KB / block (WGP has 320 KB)
  int lane = threadIdx.x & 31, w = threadIdx.x >> 5;
  int base = (blockIdx.x * 4 + w) * 32;
  h16* A = sm[w];

  __builtin_prefetch(pWe1, 0, 3);  // global_prefetch of hot weights

  // Stage concat[x_i | x_j | eemb] tile [32 x 384] f16 with b128 chunks.
#pragma unroll 8
  for (int t = 0; t < 48; ++t) {
    int c = lane + 32 * t;               // 0..1535
    int row = c / 48, c48 = c % 48;
    int seg = c48 >> 4, off = (c48 & 15) * 8;
    int e = base + row; if (e >= E) e = E - 1;
    const h16* src;
    if (seg == 0)      src = xh   + (size_t)ei[e] * HDIM + off;
    else if (seg == 1) src = xh   + (size_t)ej[e] * HDIM + off;
    else               src = eemb + (size_t)e     * HDIM + off;
    *(uint4*)(A + row * 384 + seg * 128 + off) = *(const uint4*)src;
  }
  __syncthreads();

  // GEMM1: [32,384] @ [384,128] -> 12 k-steps x 8 n-tiles x 2 M-tiles
  v8f acc[2][8] = {};
#pragma unroll
  for (int kt = 0; kt < 12; ++kt) {
    v16h a0 = load_a_frag(A,            384, kt, lane);
    v16h a1 = load_a_frag(A + 16 * 384, 384, kt, lane);
#pragma unroll
    for (int nt = 0; nt < 8; ++nt) {
      v16h b = load_b_frag(pWe1, 8, kt, nt, lane);
      acc[0][nt] = wmma16(a0, b, acc[0][nt]);
      acc[1][nt] = wmma16(a1, b, acc[1][nt]);
    }
  }
  ln_elu(acc[0], g1, b1, lane);
  ln_elu(acc[1], g1, b1, lane);

  __syncthreads();
  c_to_lds(acc[0], A,            128, lane);
  c_to_lds(acc[1], A + 16 * 128, 128, lane);
  __syncthreads();

  // GEMM2: [32,128] @ [128,128]
  v8f acc2[2][8] = {};
#pragma unroll
  for (int kt = 0; kt < 4; ++kt) {
    v16h a0 = load_a_frag(A,            128, kt, lane);
    v16h a1 = load_a_frag(A + 16 * 128, 128, kt, lane);
#pragma unroll
    for (int nt = 0; nt < 8; ++nt) {
      v16h b = load_b_frag(pWe2, 8, kt, nt, lane);
      acc2[0][nt] = wmma16(a0, b, acc2[0][nt]);
      acc2[1][nt] = wmma16(a1, b, acc2[1][nt]);
    }
  }
  ln_elu(acc2[0], g2, b2, lane);
  ln_elu(acc2[1], g2, b2, lane);

  // Attention gate + scatter
  int col = lane & 15, mofs = (lane >> 4) << 3;
  float wav[8];
#pragma unroll
  for (int nt = 0; nt < 8; ++nt) wav[nt] = wa[nt * 16 + col];
  float bav = ba[0];

#pragma unroll
  for (int half = 0; half < 2; ++half) {
#pragma unroll
    for (int r = 0; r < 8; ++r) {
      float p = 0.f;
#pragma unroll
      for (int nt = 0; nt < 8; ++nt) p += acc2[half][nt][r] * wav[nt];
#pragma unroll
      for (int m = 1; m <= 8; m <<= 1) p += __shfl_xor(p, m, 32);
      float gv = p + bav;
      int e = base + half * 16 + r + mofs;
      if (e < E) {
        int node = ei[e];
#pragma unroll
        for (int nt = 0; nt < 8; ++nt)
          atomicAdd(&macc[(size_t)node * HDIM + nt * 16 + col],
                    acc2[half][nt][r] * gv);
      }
    }
  }
}

// ---------------------------------------------------------------------------
// Node MLP: hh = elu(LN(concat[x, macc/cnt] @ Wn1)); hh = elu(LN(hh @ Wn2));
// x = hh + x.  Also re-zeroes macc for the next layer.
// ---------------------------------------------------------------------------
__global__ __launch_bounds__(128) void node_mlp_kernel(
    h16* __restrict__ xh, float* __restrict__ xf,
    float* __restrict__ macc, const float* __restrict__ cnt,
    const h16* __restrict__ pWn1, const h16* __restrict__ pWn2,
    const float* __restrict__ g1, const float* __restrict__ b1,
    const float* __restrict__ g2, const float* __restrict__ b2, int N) {
  __shared__ h16 sm[4][16 * 256];
  int lane = threadIdx.x & 31, w = threadIdx.x >> 5;
  int base = (blockIdx.x * 4 + w) * 16;
  h16* A = sm[w];

  // Stage [x_f16 | macc/cnt] tile [16 x 256] f16; zero macc behind us.
#pragma unroll
  for (int t = 0; t < 8; ++t) {
    int c = lane + 32 * t;               // 0..255: x half
    int row = c >> 4, off = (c & 15) * 8;
    int n = base + row; if (n >= N) n = N - 1;
    *(uint4*)(A + row * 256 + off) = *(const uint4*)(xh + (size_t)n * HDIM + off);
  }
#pragma unroll
  for (int t = 0; t < 16; ++t) {
    int c = lane + 32 * t;               // 0..511: m half (4 floats per chunk)
    int row = c >> 5, off4 = (c & 31) * 4;
    int n = base + row; bool ok = n < N; if (!ok) n = N - 1;
    float rc = 1.f / cnt[n];
    float4 v = *(const float4*)(macc + (size_t)n * HDIM + off4);
    h16* d = A + row * 256 + 128 + off4;
    d[0] = (h16)(v.x * rc); d[1] = (h16)(v.y * rc);
    d[2] = (h16)(v.z * rc); d[3] = (h16)(v.w * rc);
    if (ok) {
      float4 z = make_float4(0.f, 0.f, 0.f, 0.f);
      *(float4*)(macc + (size_t)n * HDIM + off4) = z;
    }
  }
  __syncthreads();

  // GEMM1: [16,256] @ [256,128]
  v8f acc[8] = {};
#pragma unroll
  for (int kt = 0; kt < 8; ++kt) {
    v16h a = load_a_frag(A, 256, kt, lane);
#pragma unroll
    for (int nt = 0; nt < 8; ++nt)
      acc[nt] = wmma16(a, load_b_frag(pWn1, 8, kt, nt, lane), acc[nt]);
  }
  ln_elu(acc, g1, b1, lane);

  __syncthreads();
  c_to_lds(acc, A, 128, lane);
  __syncthreads();

  // GEMM2: [16,128] @ [128,128]
  v8f acc2[8] = {};
#pragma unroll
  for (int kt = 0; kt < 4; ++kt) {
    v16h a = load_a_frag(A, 128, kt, lane);
#pragma unroll
    for (int nt = 0; nt < 8; ++nt)
      acc2[nt] = wmma16(a, load_b_frag(pWn2, 8, kt, nt, lane), acc2[nt]);
  }
  ln_elu(acc2, g2, b2, lane);

  // Residual and writeback (f32 + f16 shadow)
  int col = lane & 15, mofs = (lane >> 4) << 3;
#pragma unroll
  for (int nt = 0; nt < 8; ++nt)
#pragma unroll
    for (int r = 0; r < 8; ++r) {
      int n = base + r + mofs;
      if (n < N) {
        size_t idx = (size_t)n * HDIM + nt * 16 + col;
        float v = acc2[nt][r] + xf[idx];
        xf[idx] = v;
        xh[idx] = (h16)v;
      }
    }
}

// ---------------------------------------------------------------------------
// Pooling + heads (bandwidth-trivial tail)
// ---------------------------------------------------------------------------
__global__ void pool_kernel(const float* __restrict__ xf,
                            const int* __restrict__ batch,
                            float* __restrict__ gsum, int N) {
  int t = blockIdx.x * blockDim.x + threadIdx.x;
  if (t >= N * HDIM) return;
  atomicAdd(&gsum[(size_t)batch[t >> 7] * HDIM + (t & 127)], xf[t]);
}

__global__ void xg_kernel(float* gsum, const float* gcnt, int G) {
  int t = blockIdx.x * blockDim.x + threadIdx.x;
  if (t < G * HDIM) gsum[t] /= fmaxf(gcnt[t >> 7], 1.f);
}

__global__ void out_kernel(const float* __restrict__ xg,
                           const float* __restrict__ Wout,
                           const float* __restrict__ bout,
                           float* __restrict__ out, int G) {
  int g = blockIdx.x * blockDim.x + threadIdx.x;
  if (g >= G) return;
  float s = bout[0];
  for (int h = 0; h < HDIM; ++h) s += xg[(size_t)g * HDIM + h] * Wout[h];
  out[g] = s;
}

__global__ void head1_kernel(const float* __restrict__ xg,
                             const float* __restrict__ pres,
                             const float* __restrict__ outg,
                             const float* __restrict__ Wi1,
                             const float* __restrict__ bi1,
                             float* __restrict__ buf, int G, int P) {
  int t = blockIdx.x * blockDim.x + threadIdx.x;
  if (t >= G * P * HDIM) return;
  int hcol = t & 127;
  int gp = t >> 7;
  int p = gp % P, g = gp / P;
  float s = bi1[hcol];
  for (int k = 0; k < HDIM; ++k) s += xg[(size_t)g * HDIM + k] * Wi1[k * HDIM + hcol];
  s += pres[p] * Wi1[128 * HDIM + hcol];
  s += outg[g] * Wi1[129 * HDIM + hcol];
  buf[t] = elu_f(s);
}

__global__ void head2_kernel(const float* __restrict__ buf,
                             const float* __restrict__ Wi2,
                             const float* __restrict__ bi2,
                             float* __restrict__ q, int GP) {
  int t = blockIdx.x * blockDim.x + threadIdx.x;
  if (t >= GP) return;
  float s = bi2[0];
  for (int h = 0; h < HDIM; ++h) s += buf[(size_t)t * HDIM + h] * Wi2[h];
  q[t] = fmaxf(s, 0.f) + log1pf(expf(-fabsf(s)));  // stable softplus
}

// ---------------------------------------------------------------------------
// Host-side orchestration
// ---------------------------------------------------------------------------
extern "C" void kernel_launch(void* const* d_in, const int* in_sizes, int n_in,
                              void* d_out, int out_size, void* d_ws, size_t ws_size,
                              hipStream_t stream) {
  const float* x      = (const float*)d_in[0];
  const float* ea     = (const float*)d_in[1];
  const float* pres   = (const float*)d_in[2];
  const int*   eidx   = (const int*)  d_in[3];
  const int*   batch  = (const int*)  d_in[4];
  const float* Wemb   = (const float*)d_in[5];
  const float* Wee    = (const float*)d_in[6];
  const float* bee    = (const float*)d_in[7];
  const float* We1    = (const float*)d_in[8];
  const float* We2    = (const float*)d_in[9];
  const float* eg1    = (const float*)d_in[10];
  const float* eb1    = (const float*)d_in[11];
  const float* eg2    = (const float*)d_in[12];
  const float* eb2    = (const float*)d_in[13];
  const float* Wn1    = (const float*)d_in[14];
  const float* Wn2    = (const float*)d_in[15];
  const float* ng1    = (const float*)d_in[16];
  const float* nb1    = (const float*)d_in[17];
  const float* ng2    = (const float*)d_in[18];
  const float* nb2    = (const float*)d_in[19];
  const float* Wa     = (const float*)d_in[20];
  const float* ba     = (const float*)d_in[21];
  const float* Wout   = (const float*)d_in[22];
  const float* bout   = (const float*)d_in[23];
  const float* Wi1    = (const float*)d_in[24];
  const float* bi1    = (const float*)d_in[25];
  const float* Wi2    = (const float*)d_in[26];
  const float* bi2    = (const float*)d_in[27];

  const int N = in_sizes[0];
  const int E = in_sizes[1];
  const int P = in_sizes[2];
  const int L = in_sizes[8] / (3 * HDIM * HDIM);
  const int G = out_size / (1 + P);

  const int* ei = eidx;
  const int* ej = eidx + E;

  // Workspace carve-out
  char* ws = (char*)d_ws;
  size_t cur = 0;
  auto carve = [&](size_t bytes) -> void* {
    void* p = ws + cur;
    cur += (bytes + 255) & ~(size_t)255;
    return p;
  };
  float* xf      = (float*)carve((size_t)N * HDIM * 4);
  h16*   xh      = (h16*)  carve((size_t)N * HDIM * 2);
  h16*   eemb    = (h16*)  carve((size_t)E * HDIM * 2);
  float* macc    = (float*)carve((size_t)N * HDIM * 4);
  float* cnt     = (float*)carve((size_t)N * 4);
  float* gcnt    = (float*)carve((size_t)G * 4);
  float* gsum    = (float*)carve((size_t)G * HDIM * 4);
  float* headbuf = (float*)carve((size_t)G * P * HDIM * 4);
  const size_t FR = 512;                 // halves per fragment (32 lanes x 16)
  h16* pWee = (h16*)carve(2  * 8 * FR * 2);
  h16* pWe1 = (h16*)carve((size_t)L * 12 * 8 * FR * 2);
  h16* pWe2 = (h16*)carve((size_t)L *  4 * 8 * FR * 2);
  h16* pWn1 = (h16*)carve((size_t)L *  8 * 8 * FR * 2);
  h16* pWn2 = (h16*)carve((size_t)L *  4 * 8 * FR * 2);
  (void)ws_size; (void)n_in;

  auto cdiv = [](int a, int b) { return (a + b - 1) / b; };

  // 1) Zero accumulators
  zero_f32_kernel<<<cdiv(N * HDIM, 256), 256, 0, stream>>>(macc, N * HDIM);
  zero_f32_kernel<<<cdiv(N, 256), 256, 0, stream>>>(cnt, N);
  zero_f32_kernel<<<cdiv(G, 256), 256, 0, stream>>>(gcnt, G);
  zero_f32_kernel<<<cdiv(G * HDIM, 256), 256, 0, stream>>>(gsum, G * HDIM);

  // 2) h0 = x @ W_emb (f32 + f16 shadow)
  embed_kernel<<<cdiv(N * HDIM, 256), 256, 0, stream>>>(x, Wemb, xf, xh, N);

  // 3) degree counts
  count_edge_kernel<<<cdiv(E, 256), 256, 0, stream>>>(ei, cnt, E);
  clamp1_kernel<<<cdiv(N, 256), 256, 0, stream>>>(cnt, N);
  count_node_kernel<<<cdiv(N, 256), 256, 0, stream>>>(batch, gcnt, N);

  // 4) pack weights into WMMA B-fragment layout (f16)
  pack_b_kernel<<<cdiv(2 * 8 * 512, 256), 256, 0, stream>>>(Wee, pWee, 2, 8, HDIM);
  for (int l = 0; l < L; ++l) {
    pack_b_kernel<<<cdiv(12 * 8 * 512, 256), 256, 0, stream>>>(
        We1 + (size_t)l * 384 * HDIM, pWe1 + (size_t)l * 12 * 8 * FR, 12, 8, HDIM);
    pack_b_kernel<<<cdiv(4 * 8 * 512, 256), 256, 0, stream>>>(
        We2 + (size_t)l * HDIM * HDIM, pWe2 + (size_t)l * 4 * 8 * FR, 4, 8, HDIM);
    pack_b_kernel<<<cdiv(8 * 8 * 512, 256), 256, 0, stream>>>(
        Wn1 + (size_t)l * 256 * HDIM, pWn1 + (size_t)l * 8 * 8 * FR, 8, 8, HDIM);
    pack_b_kernel<<<cdiv(4 * 8 * 512, 256), 256, 0, stream>>>(
        Wn2 + (size_t)l * HDIM * HDIM, pWn2 + (size_t)l * 4 * 8 * FR, 4, 8, HDIM);
  }

  // 5) edge embedding (WMMA)
  int etiles16 = cdiv(E, 16);
  rbf_edge_kernel<<<cdiv(etiles16, 4), 128, 0, stream>>>(ea, pWee, bee, eemb, E);

  // 6) message-passing layers (WMMA edge MLP + WMMA node MLP)
  int etiles32 = cdiv(E, 32);
  int ntiles   = cdiv(N, 16);
  for (int l = 0; l < L; ++l) {
    edge_mlp_kernel<<<cdiv(etiles32, 4), 128, 0, stream>>>(
        ei, ej, xh, eemb,
        pWe1 + (size_t)l * 12 * 8 * FR, pWe2 + (size_t)l * 4 * 8 * FR,
        eg1 + l * HDIM, eb1 + l * HDIM, eg2 + l * HDIM, eb2 + l * HDIM,
        Wa + l * HDIM, ba + l, macc, E);
    node_mlp_kernel<<<cdiv(ntiles, 4), 128, 0, stream>>>(
        xh, xf, macc, cnt,
        pWn1 + (size_t)l * 8 * 8 * FR, pWn2 + (size_t)l * 4 * 8 * FR,
        ng1 + l * HDIM, nb1 + l * HDIM, ng2 + l * HDIM, nb2 + l * HDIM, N);
  }

  // 7) graph pooling + linear head -> d_out[0..G)
  float* outp = (float*)d_out;
  pool_kernel<<<cdiv(N * HDIM, 256), 256, 0, stream>>>(xf, batch, gsum, N);
  xg_kernel<<<cdiv(G * HDIM, 256), 256, 0, stream>>>(gsum, gcnt, G);
  out_kernel<<<cdiv(G, 256), 256, 0, stream>>>(gsum, Wout, bout, outp, G);

  // 8) inference head -> d_out[G .. G+G*P)
  head1_kernel<<<cdiv(G * P * HDIM, 256), 256, 0, stream>>>(
      gsum, pres, outp, Wi1, bi1, headbuf, G, P);
  head2_kernel<<<cdiv(G * P, 256), 256, 0, stream>>>(
      headbuf, Wi2, bi2, outp + G, G * P);
}
